// LPModel_75694503624797
// MI455X (gfx1250) — matrix operand order repaired and checked by reading, not compiled, not run on hardware
//
#include <hip/hip_runtime.h>
#include <stdint.h>

// ---------------------------------------------------------------------------
// Types for CDNA5 WMMA (wave32)
// ---------------------------------------------------------------------------
typedef __bf16 bf16;
typedef __attribute__((ext_vector_type(16))) __bf16 v16bf;
typedef __attribute__((ext_vector_type(8)))  __bf16 v8bf;
typedef __attribute__((ext_vector_type(4)))  __bf16 v4bf;
typedef __attribute__((ext_vector_type(8)))  float  v8f;

// ---------------------------------------------------------------------------
// fp32 -> bf16 conversion
// ---------------------------------------------------------------------------
__global__ void f32_to_bf16_kernel(const float* __restrict__ x,
                                   bf16* __restrict__ y, long long n) {
    long long i = (long long)blockIdx.x * blockDim.x + threadIdx.x;
    if (i < n) y[i] = (bf16)x[i];
}

// ---------------------------------------------------------------------------
// LayerNorm over 128-wide rows, output bf16. One wave32 per row, 4 elems/lane.
// ---------------------------------------------------------------------------
__global__ __launch_bounds__(256) void ln_bf16_kernel(
        const float* __restrict__ x, const float* __restrict__ g,
        const float* __restrict__ b, bf16* __restrict__ y, int R) {
    int row  = blockIdx.x * 8 + (threadIdx.x >> 5);
    int lane = threadIdx.x & 31;
    if (row >= R) return;
    const float* rp = x + (size_t)row * 128;
    float4 v = *(const float4*)(rp + lane * 4);
    float s = v.x + v.y + v.z + v.w;
    #pragma unroll
    for (int o = 16; o > 0; o >>= 1) s += __shfl_xor(s, o, 32);
    float mean = s * (1.0f / 128.0f);
    float dx = v.x - mean, dy = v.y - mean, dz = v.z - mean, dw = v.w - mean;
    float vs = dx * dx + dy * dy + dz * dz + dw * dw;
    #pragma unroll
    for (int o = 16; o > 0; o >>= 1) vs += __shfl_xor(vs, o, 32);
    float rstd = rsqrtf(vs * (1.0f / 128.0f) + 1e-5f);
    int c = lane * 4;
    v4bf o4;
    o4[0] = (bf16)(dx * rstd * g[c + 0] + b[c + 0]);
    o4[1] = (bf16)(dy * rstd * g[c + 1] + b[c + 1]);
    o4[2] = (bf16)(dz * rstd * g[c + 2] + b[c + 2]);
    o4[3] = (bf16)(dw * rstd * g[c + 3] + b[c + 3]);
    *(v4bf*)(y + (size_t)row * 128 + c) = o4;
}

// ---------------------------------------------------------------------------
// WMMA bf16 GEMM:  C[R,Nout] = A[R,K] * W[Nout,K]^T + bias [+ReLU][+resid]
// Register-blocked: one wave computes a 16x64 output tile (4 accumulators),
// loading each A-tile once per K-step and reusing it across 4 B-tiles.
// R mult of 16, K mult of 32, Nout mult of 64.
// A-tile layout in contiguous K order, K = 16*half + s (lane%16 = M row, K runs per half).
// B-tile layout in the same contiguous K order as A (lane%16 = N col, contiguous K/half);
// since B[k][n] = W[n][k], each lane streams W row (col) contiguously.
// ---------------------------------------------------------------------------
__device__ inline v16bf load_a_tile(const bf16* base, int half) {
    union { v16bf v; v8bf h[2]; } u;
    u.h[0] = *(const v8bf*)(base + half * 16);       // K = h*16 .. h*16+7
    u.h[1] = *(const v8bf*)(base + half * 16 + 8);   // K = h*16+8 .. h*16+15
    return u.v;
}
__device__ inline v16bf load_b_tile(const bf16* base, int half) {
    union { v16bf v; v8bf h[2]; } u;
    const bf16* p = base + half * 16;                // K = h*16 .. h*16+15
    u.h[0] = *(const v8bf*)(p);
    u.h[1] = *(const v8bf*)(p + 8);
    return u.v;
}

template <bool RELU, bool BF16OUT, bool RESID>
__global__ __launch_bounds__(256) void gemm_wmma(
        const bf16* __restrict__ A, const bf16* __restrict__ W,
        const float* __restrict__ bias, const float* __restrict__ resid,
        float* __restrict__ Cf, bf16* __restrict__ Cb,
        int R, int K, int Nout) {
    int wave = blockIdx.x * 8 + (threadIdx.x >> 5);
    int lane = threadIdx.x & 31;
    int colGrp = Nout >> 6;                          // groups of 64 columns
    int nTiles = (R >> 4) * colGrp;
    if (wave >= nTiles) return;                      // wave-uniform branch
    int rt = wave / colGrp, cg = wave - rt * colGrp;
    int rl = lane & 15, half = lane >> 4;

    const bf16* Ap  = A + (size_t)(rt * 16 + rl) * K;
    const bf16* Wp0 = W + (size_t)(cg * 64 + 0  + rl) * K;
    const bf16* Wp1 = W + (size_t)(cg * 64 + 16 + rl) * K;
    const bf16* Wp2 = W + (size_t)(cg * 64 + 32 + rl) * K;
    const bf16* Wp3 = W + (size_t)(cg * 64 + 48 + rl) * K;

    v8f acc0 = {}, acc1 = {}, acc2 = {}, acc3 = {};
    for (int kk = 0; kk < K; kk += 32) {
        v16bf a  = load_a_tile(Ap  + kk, half);
        v16bf b0 = load_b_tile(Wp0 + kk, half);
        v16bf b1 = load_b_tile(Wp1 + kk, half);
        v16bf b2 = load_b_tile(Wp2 + kk, half);
        v16bf b3 = load_b_tile(Wp3 + kk, half);
        acc0 = __builtin_amdgcn_wmma_f32_16x16x32_bf16(false, a, false, b0, (short)0, acc0, false, false);
        acc1 = __builtin_amdgcn_wmma_f32_16x16x32_bf16(false, a, false, b1, (short)0, acc1, false, false);
        acc2 = __builtin_amdgcn_wmma_f32_16x16x32_bf16(false, a, false, b2, (short)0, acc2, false, false);
        acc3 = __builtin_amdgcn_wmma_f32_16x16x32_bf16(false, a, false, b3, (short)0, acc3, false, false);
    }

    #pragma unroll
    for (int sub = 0; sub < 4; ++sub) {
        v8f acc = sub == 0 ? acc0 : sub == 1 ? acc1 : sub == 2 ? acc2 : acc3;
        int col = cg * 64 + sub * 16 + rl;
        float bv = bias[col];
        #pragma unroll
        for (int r = 0; r < 8; ++r) {
            int row = rt * 16 + half * 8 + r;
            float v = acc[r] + bv;
            if (RELU)  v = v > 0.0f ? v : 0.0f;
            if (RESID) v += resid[(size_t)row * Nout + col];
            if (BF16OUT) Cb[(size_t)row * Nout + col] = (bf16)v;
            else         Cf[(size_t)row * Nout + col] = v;
        }
    }
}

// ---------------------------------------------------------------------------
// Segment softmax helpers (atomics). Monotonic float<->uint for atomicMax.
// ---------------------------------------------------------------------------
__device__ inline unsigned ord_enc(float f) {
    unsigned u = __float_as_uint(f);
    return (u & 0x80000000u) ? ~u : (u | 0x80000000u);
}
__device__ inline float ord_dec(unsigned u) {
    unsigned bits = (u & 0x80000000u) ? (u ^ 0x80000000u) : ~u;
    return __uint_as_float(bits);
}

// --- node-MHA: score[e,h] = <eQ[e]+nQ[src], eK[e]+nK[dst]>_h * 0.25 --------
__global__ void score_node_kernel(const bf16* __restrict__ eQ, const bf16* __restrict__ eK,
                                  const bf16* __restrict__ nQ, const bf16* __restrict__ nK,
                                  const int* __restrict__ src, const int* __restrict__ dst,
                                  float* __restrict__ score, unsigned* __restrict__ segmax,
                                  int M) {
    long long idx = (long long)blockIdx.x * blockDim.x + threadIdx.x;
    if (idx >= (long long)M * 8) return;
    int e = (int)(idx >> 3), h = (int)(idx & 7);
    int s = src[e], d = dst[e];
    const bf16* q0 = eQ + (size_t)e * 128 + h * 16;
    const bf16* q1 = nQ + (size_t)s * 128 + h * 16;
    const bf16* k0 = eK + (size_t)e * 128 + h * 16;
    const bf16* k1 = nK + (size_t)d * 128 + h * 16;
    float acc = 0.0f;
    #pragma unroll
    for (int i = 0; i < 16; ++i) {
        float q = (float)q0[i] + (float)q1[i];
        float k = (float)k0[i] + (float)k1[i];
        acc += q * k;
    }
    acc *= 0.25f;
    score[idx] = acc;
    atomicMax(segmax + (size_t)s * 8 + h, ord_enc(acc));
}

// --- edge-MHA: 3-term Q/K over pair lists ----------------------------------
__global__ void score_edge_kernel(const bf16* __restrict__ eQ, const bf16* __restrict__ eK,
                                  const bf16* __restrict__ nQ, const bf16* __restrict__ nK,
                                  const int* __restrict__ src, const int* __restrict__ dst,
                                  const int* __restrict__ pi, const int* __restrict__ pj,
                                  float* __restrict__ score, unsigned* __restrict__ segmax,
                                  int P) {
    long long idx = (long long)blockIdx.x * blockDim.x + threadIdx.x;
    if (idx >= (long long)P * 8) return;
    int p = (int)(idx >> 3), h = (int)(idx & 7);
    int i = pi[p], j = pj[p];
    int si = src[i], di = dst[i], sj = src[j], dj = dst[j];
    const bf16* q0 = eQ + (size_t)i * 128 + h * 16;
    const bf16* q1 = nQ + (size_t)si * 128 + h * 16;
    const bf16* q2 = nQ + (size_t)di * 128 + h * 16;
    const bf16* k0 = eK + (size_t)j * 128 + h * 16;
    const bf16* k1 = nK + (size_t)sj * 128 + h * 16;
    const bf16* k2 = nK + (size_t)dj * 128 + h * 16;
    float acc = 0.0f;
    #pragma unroll
    for (int t = 0; t < 16; ++t) {
        float q = (float)q0[t] + (float)q1[t] + (float)q2[t];
        float k = (float)k0[t] + (float)k1[t] + (float)k2[t];
        acc += q * k;
    }
    acc *= 0.25f;
    score[idx] = acc;
    atomicMax(segmax + (size_t)i * 8 + h, ord_enc(acc));
}

// --- shared: w = exp(score - segmax[seg]); denom[seg] += w (in place) ------
__global__ void softmax_expsum_kernel(const int* __restrict__ seg,
                                      float* __restrict__ score,
                                      const unsigned* __restrict__ segmax,
                                      float* __restrict__ denom, int E) {
    long long idx = (long long)blockIdx.x * blockDim.x + threadIdx.x;
    if (idx >= (long long)E * 8) return;
    int e = (int)(idx >> 3), h = (int)(idx & 7);
    int sg = seg[e];
    float mx = ord_dec(segmax[(size_t)sg * 8 + h]);
    float w = __expf(score[idx] - mx);
    score[idx] = w;
    atomicAdd(denom + (size_t)sg * 8 + h, w);
}

// --- node-MHA: out[src] += (w/denom) * (eV[e] + nV[dst]) -------------------
__global__ void accum_node_kernel(const float* __restrict__ w, const float* __restrict__ denom,
                                  const bf16* __restrict__ eV, const bf16* __restrict__ nV,
                                  const int* __restrict__ src, const int* __restrict__ dst,
                                  float* __restrict__ out, int M) {
    long long idx = (long long)blockIdx.x * blockDim.x + threadIdx.x;
    if (idx >= (long long)M * 128) return;
    int e = (int)(idx >> 7), d = (int)(idx & 127), h = d >> 4;
    int s = src[e], t = dst[e];
    float ww = w[(size_t)e * 8 + h] / denom[(size_t)s * 8 + h];
    float v = (float)eV[(size_t)e * 128 + d] + (float)nV[(size_t)t * 128 + d];
    atomicAdd(out + (size_t)s * 128 + d, ww * v);
}

// --- edge-MHA: out[pi] += (w/denom) * (eV[pj] + nV[src[pj]] + nV[dst[pj]]) -
__global__ void accum_edge_kernel(const float* __restrict__ w, const float* __restrict__ denom,
                                  const bf16* __restrict__ eV, const bf16* __restrict__ nV,
                                  const int* __restrict__ src, const int* __restrict__ dst,
                                  const int* __restrict__ pi, const int* __restrict__ pj,
                                  float* __restrict__ out, int P) {
    long long idx = (long long)blockIdx.x * blockDim.x + threadIdx.x;
    if (idx >= (long long)P * 128) return;
    int p = (int)(idx >> 7), d = (int)(idx & 127), h = d >> 4;
    int i = pi[p], j = pj[p];
    float ww = w[(size_t)p * 8 + h] / denom[(size_t)i * 8 + h];
    float v = (float)eV[(size_t)j * 128 + d]
            + (float)nV[(size_t)src[j] * 128 + d]
            + (float)nV[(size_t)dst[j] * 128 + d];
    atomicAdd(out + (size_t)i * 128 + d, ww * v);
}

// ---------------------------------------------------------------------------
// Host side
// ---------------------------------------------------------------------------
static inline int nblk(long long work, int per) { return (int)((work + per - 1) / per); }

extern "C" void kernel_launch(void* const* d_in, const int* in_sizes, int n_in,
                              void* d_out, int out_size, void* d_ws, size_t ws_size,
                              hipStream_t stream) {
    const int D = 128, HID = 512;
    const float* node = (const float*)d_in[0];
    const float* edge = (const float*)d_in[1];
    const int*   eidx = (const int*)d_in[2];
    const int*   prs  = (const int*)d_in[3];
    const int N = in_sizes[0] / D;
    const int M = in_sizes[2] / 2;
    const int P = in_sizes[3] / 2;
    const int* src = eidx;
    const int* dst = eidx + M;
    const int* pi  = prs;
    const int* pj  = prs + P;

    // ---- flattened params (JAX pytree order: dicts sorted by key) ----
    enum {
        EF1_W = 4, EF1_B, EF2_W, EF2_B,
        E_WEK_W = 8, E_WEK_B, E_WEQ_W, E_WEQ_B, E_WEV_W, E_WEV_B,
        E_WNK_W, E_WNK_B, E_WNQ_W, E_WNQ_B, E_WNV_W, E_WNV_B, E_OUT_W, E_OUT_B,
        LN1E_G = 22, LN1E_B, LN1N_G, LN1N_B, LN2E_G, LN2E_B, LN2N_G, LN2N_B,
        LNEF_G, LNEF_B, LNNF_G, LNNF_B,
        NF1_W = 34, NF1_B, NF2_W, NF2_B,
        N_WEK_W = 38, N_WEK_B, N_WEQ_W, N_WEQ_B, N_WEV_W, N_WEV_B,
        N_WNK_W, N_WNK_B, N_WNQ_W, N_WNQ_B, N_WNV_W, N_WNV_B, N_OUT_W, N_OUT_B
    };
    auto F = [&](int i) { return (const float*)d_in[i]; };

    // ---- workspace bump allocator ----
    char*  ws  = (char*)d_ws;
    size_t off = 0;
    auto alloc = [&](size_t bytes) -> void* {
        void* r = ws + off;
        off = (off + bytes + 255) & ~(size_t)255;
        return r;
    };

    // ---- bf16 weight copies ----
    const int wIdx[18] = { N_WNQ_W, N_WNK_W, N_WNV_W, N_WEQ_W, N_WEK_W, N_WEV_W, N_OUT_W,
                           E_WNQ_W, E_WNK_W, E_WNV_W, E_WEQ_W, E_WEK_W, E_WEV_W, E_OUT_W,
                           NF1_W, NF2_W, EF1_W, EF2_W };
    const long long wEl[18] = { 16384,16384,16384,16384,16384,16384,16384,
                                16384,16384,16384,16384,16384,16384,16384,
                                65536,65536,65536,65536 };
    bf16* wb[18];
    for (int i = 0; i < 18; ++i) {
        wb[i] = (bf16*)alloc((size_t)wEl[i] * 2);
        f32_to_bf16_kernel<<<nblk(wEl[i], 256), 256, 0, stream>>>(F(wIdx[i]), wb[i], wEl[i]);
    }
    bf16 *nWnq = wb[0], *nWnk = wb[1], *nWnv = wb[2], *nWeq = wb[3], *nWek = wb[4],
         *nWev = wb[5], *nWout = wb[6];
    bf16 *eWnq = wb[7], *eWnk = wb[8], *eWnv = wb[9], *eWeq = wb[10], *eWek = wb[11],
         *eWev = wb[12], *eWout = wb[13];
    bf16 *wnf1 = wb[14], *wnf2 = wb[15], *wef1 = wb[16], *wef2 = wb[17];

    // ---- activation buffers ----
    const int mxNM = (N > M) ? N : M;
    bf16*  lnN    = (bf16*)alloc((size_t)N * D * 2);
    bf16*  lnE    = (bf16*)alloc((size_t)M * D * 2);
    bf16*  pNq    = (bf16*)alloc((size_t)N * D * 2);
    bf16*  pNk    = (bf16*)alloc((size_t)N * D * 2);
    bf16*  pNv    = (bf16*)alloc((size_t)N * D * 2);
    bf16*  pEq    = (bf16*)alloc((size_t)M * D * 2);
    bf16*  pEk    = (bf16*)alloc((size_t)M * D * 2);
    bf16*  pEv    = (bf16*)alloc((size_t)M * D * 2);
    float* attn   = (float*)alloc((size_t)M * D * 4);
    bf16*  attnb  = (bf16*)alloc((size_t)M * D * 2);
    float* score  = (float*)alloc((size_t)P * 8 * 4);
    unsigned* smax = (unsigned*)alloc((size_t)mxNM * 8 * 4);
    float* denom  = (float*)alloc((size_t)mxNM * 8 * 4);
    float* n1     = (float*)alloc((size_t)N * D * 4);
    float* e1     = (float*)alloc((size_t)M * D * 4);
    bf16*  hid    = (bf16*)alloc((size_t)M * HID * 2);
    float* n2     = (float*)d_out;
    float* e2     = (float*)d_out + (size_t)N * D;

    auto gemmTiles = [](int R, int Nout) { return (long long)(R >> 4) * (Nout >> 6); };

    // =============== node MHA =================================================
    ln_bf16_kernel<<<nblk(N, 8), 256, 0, stream>>>(node, F(LN1N_G), F(LN1N_B), lnN, N);
    ln_bf16_kernel<<<nblk(M, 8), 256, 0, stream>>>(edge, F(LN1E_G), F(LN1E_B), lnE, M);

    gemm_wmma<false, true, false><<<nblk(gemmTiles(N, D), 8), 256, 0, stream>>>(
        lnN, nWnq, F(N_WNQ_B), nullptr, nullptr, pNq, N, D, D);
    gemm_wmma<false, true, false><<<nblk(gemmTiles(N, D), 8), 256, 0, stream>>>(
        lnN, nWnk, F(N_WNK_B), nullptr, nullptr, pNk, N, D, D);
    gemm_wmma<false, true, false><<<nblk(gemmTiles(N, D), 8), 256, 0, stream>>>(
        lnN, nWnv, F(N_WNV_B), nullptr, nullptr, pNv, N, D, D);
    gemm_wmma<false, true, false><<<nblk(gemmTiles(M, D), 8), 256, 0, stream>>>(
        lnE, nWeq, F(N_WEQ_B), nullptr, nullptr, pEq, M, D, D);
    gemm_wmma<false, true, false><<<nblk(gemmTiles(M, D), 8), 256, 0, stream>>>(
        lnE, nWek, F(N_WEK_B), nullptr, nullptr, pEk, M, D, D);
    gemm_wmma<false, true, false><<<nblk(gemmTiles(M, D), 8), 256, 0, stream>>>(
        lnE, nWev, F(N_WEV_B), nullptr, nullptr, pEv, M, D, D);

    hipMemsetAsync(smax,  0, (size_t)N * 8 * 4, stream);
    hipMemsetAsync(denom, 0, (size_t)N * 8 * 4, stream);
    hipMemsetAsync(attn,  0, (size_t)N * D * 4, stream);
    score_node_kernel<<<nblk((long long)M * 8, 256), 256, 0, stream>>>(
        pEq, pEk, pNq, pNk, src, dst, score, smax, M);
    softmax_expsum_kernel<<<nblk((long long)M * 8, 256), 256, 0, stream>>>(
        src, score, smax, denom, M);
    accum_node_kernel<<<nblk((long long)M * 128, 256), 256, 0, stream>>>(
        score, denom, pEv, pNv, src, dst, attn, M);

    f32_to_bf16_kernel<<<nblk((long long)N * D, 256), 256, 0, stream>>>(
        attn, attnb, (long long)N * D);
    gemm_wmma<false, false, true><<<nblk(gemmTiles(N, D), 8), 256, 0, stream>>>(
        attnb, nWout, F(N_OUT_B), node, n1, nullptr, N, D, D);   // n1 = proj + node

    // =============== node FF ==================================================
    ln_bf16_kernel<<<nblk(N, 8), 256, 0, stream>>>(n1, F(LNNF_G), F(LNNF_B), lnN, N);
    gemm_wmma<true, true, false><<<nblk(gemmTiles(N, HID), 8), 256, 0, stream>>>(
        lnN, wnf1, F(NF1_B), nullptr, nullptr, hid, N, D, HID);
    gemm_wmma<false, false, true><<<nblk(gemmTiles(N, D), 8), 256, 0, stream>>>(
        hid, wnf2, F(NF2_B), n1, n2, nullptr, N, HID, D);        // n2 -> d_out

    // =============== edge MHA =================================================
    ln_bf16_kernel<<<nblk(N, 8), 256, 0, stream>>>(n2, F(LN2N_G), F(LN2N_B), lnN, N);
    ln_bf16_kernel<<<nblk(M, 8), 256, 0, stream>>>(edge, F(LN2E_G), F(LN2E_B), lnE, M);

    gemm_wmma<false, true, false><<<nblk(gemmTiles(N, D), 8), 256, 0, stream>>>(
        lnN, eWnq, F(E_WNQ_B), nullptr, nullptr, pNq, N, D, D);
    gemm_wmma<false, true, false><<<nblk(gemmTiles(N, D), 8), 256, 0, stream>>>(
        lnN, eWnk, F(E_WNK_B), nullptr, nullptr, pNk, N, D, D);
    gemm_wmma<false, true, false><<<nblk(gemmTiles(N, D), 8), 256, 0, stream>>>(
        lnN, eWnv, F(E_WNV_B), nullptr, nullptr, pNv, N, D, D);
    gemm_wmma<false, true, false><<<nblk(gemmTiles(M, D), 8), 256, 0, stream>>>(
        lnE, eWeq, F(E_WEQ_B), nullptr, nullptr, pEq, M, D, D);
    gemm_wmma<false, true, false><<<nblk(gemmTiles(M, D), 8), 256, 0, stream>>>(
        lnE, eWek, F(E_WEK_B), nullptr, nullptr, pEk, M, D, D);
    gemm_wmma<false, true, false><<<nblk(gemmTiles(M, D), 8), 256, 0, stream>>>(
        lnE, eWev, F(E_WEV_B), nullptr, nullptr, pEv, M, D, D);

    hipMemsetAsync(smax,  0, (size_t)M * 8 * 4, stream);
    hipMemsetAsync(denom, 0, (size_t)M * 8 * 4, stream);
    hipMemsetAsync(attn,  0, (size_t)M * D * 4, stream);
    score_edge_kernel<<<nblk((long long)P * 8, 256), 256, 0, stream>>>(
        pEq, pEk, pNq, pNk, src, dst, pi, pj, score, smax, P);
    softmax_expsum_kernel<<<nblk((long long)P * 8, 256), 256, 0, stream>>>(
        pi, score, smax, denom, P);
    accum_edge_kernel<<<nblk((long long)P * 128, 256), 256, 0, stream>>>(
        score, denom, pEv, pNv, src, dst, pi, pj, attn, P);

    f32_to_bf16_kernel<<<nblk((long long)M * D, 256), 256, 0, stream>>>(
        attn, attnb, (long long)M * D);
    gemm_wmma<false, false, true><<<nblk(gemmTiles(M, D), 8), 256, 0, stream>>>(
        attnb, eWout, F(E_OUT_B), edge, e1, nullptr, M, D, D);   // e1 = proj + edge

    // =============== edge FF ==================================================
    ln_bf16_kernel<<<nblk(M, 8), 256, 0, stream>>>(e1, F(LNEF_G), F(LNEF_B), lnE, M);
    gemm_wmma<true, true, false><<<nblk(gemmTiles(M, HID), 8), 256, 0, stream>>>(
        lnE, wef1, F(EF1_B), nullptr, nullptr, hid, M, D, HID);
    gemm_wmma<false, false, true><<<nblk(gemmTiles(M, D), 8), 256, 0, stream>>>(
        hid, wef2, F(EF2_B), e1, e2, nullptr, M, HID, D);        // e2 -> d_out
}
